// CodebookWrapperLinear_69887707841172
// MI455X (gfx1250) — matrix operand order repaired and checked by reading, not compiled, not run on hardware
//
#include <hip/hip_runtime.h>
#include <hip/hip_bf16.h>

// ---------------------------------------------------------------------------
// CodebookWrapperLinear on MI455X (gfx1250, wave32, WMMA + async-LDS copies)
//   out[B,S,OUT] = x[B,S,IN] @ dequant(w)[OUT,IN]^T
//   M = B*S = 8192, K = IN = 4096, N = OUT = 16384   (~1.1 TFLOP, compute-bound)
// Pass 1: dequant 2-bit codebook weights -> bf16 (134MB), x f32 -> bf16 (67MB).
// Pass 2: bf16 WMMA GEMM, 128x128x64 tiles, LDS double-buffered, tile fill via
//         GLOBAL_LOAD_ASYNC_TO_LDS_B128 (ASYNCcnt) instead of VGPR staging.
// ---------------------------------------------------------------------------

typedef __attribute__((ext_vector_type(16))) __bf16 v16bf;
typedef __attribute__((ext_vector_type(8)))  __bf16 v8bf;
typedef __attribute__((ext_vector_type(8)))  float  v8f;

constexpr int Bsz = 4;
constexpr int Ssz = 2048;
constexpr int INsz = 4096;
constexpr int OUTsz = 16384;
constexpr int GS = 32;
constexpr int Gn = INsz / GS;          // 128 groups per out-row

constexpr int Mdim = Bsz * Ssz;        // 8192
constexpr int Ndim = OUTsz;            // 16384
constexpr int Kdim = INsz;             // 4096

constexpr int BM = 128;
constexpr int BN = 128;
constexpr int BK = 64;                 // 2 WMMA K-steps per LDS tile
constexpr int LDT = BK + 8;            // padded LDS row stride: 72 elem = 144B

// -------------------------- pass 1a: x f32 -> bf16 --------------------------
__global__ __launch_bounds__(256)
void cvt_x_kernel(const float* __restrict__ x, __bf16* __restrict__ xb) {
  const int i = blockIdx.x * 256 + threadIdx.x;      // one thread = 8 elements
  const float4* p = (const float4*)x + (size_t)i * 2;
  float4 a = p[0];
  float4 b = p[1];
  v8bf o;
  o[0] = (__bf16)a.x; o[1] = (__bf16)a.y; o[2] = (__bf16)a.z; o[3] = (__bf16)a.w;
  o[4] = (__bf16)b.x; o[5] = (__bf16)b.y; o[6] = (__bf16)b.z; o[7] = (__bf16)b.w;
  ((v8bf*)xb)[i] = o;
}

// ------------------ pass 1b: codebook dequant -> bf16 W[N,K] ----------------
__global__ __launch_bounds__(256)
void dequant_w_kernel(const float* __restrict__ codebook,
                      const float* __restrict__ scale,
                      const int*   __restrict__ indexes,
                      __bf16*      __restrict__ W) {
  const int g = blockIdx.x * 256 + threadIdx.x;      // group id in [0, OUT*G)
  const float c0 = codebook[0], c1 = codebook[1];
  const float c2 = codebook[2], c3 = codebook[3];
  float mx = fmaxf(fmaxf(fabsf(c0), fabsf(c1)), fmaxf(fabsf(c2), fabsf(c3)));
  mx = fmaxf(mx, 1e-8f);
  const float e = __expf(scale[g]) / mx;             // exp(scale) * cb_norm
  const float v0 = c0 * e, v1 = c1 * e, v2 = c2 * e, v3 = c3 * e;

  const int*  ip = indexes + (size_t)g * GS;
  __bf16*     wp = W       + (size_t)g * GS;         // groups are K-contiguous
#pragma unroll
  for (int c = 0; c < 4; ++c) {                      // 4 x 8 elements
    int4 i0 = ((const int4*)ip)[c * 2];
    int4 i1 = ((const int4*)ip)[c * 2 + 1];
    int ii[8] = {i0.x, i0.y, i0.z, i0.w, i1.x, i1.y, i1.z, i1.w};
    v8bf o;
#pragma unroll
    for (int j = 0; j < 8; ++j) {
      const int v = ii[j];
      const float f = (v < 2) ? (v == 0 ? v0 : v1) : (v == 2 ? v2 : v3);
      o[j] = (__bf16)f;
    }
    ((v8bf*)wp)[c] = o;
  }
}

// ------------------------------ pass 2: GEMM --------------------------------
// Async copy 64B (4 x b128) global -> LDS. The instruction immediate offset is
// added to BOTH the global and the LDS address (ISA 08 §4.4), so one LDS-addr
// VGPR + one global-addr VGPR pair covers the whole run. Tracked by ASYNCcnt.
__device__ __forceinline__ void async_fill64(unsigned int lds, const __bf16* g) {
  unsigned long long ga = (unsigned long long)g;
  asm volatile(
      "global_load_async_to_lds_b128 %0, %1, off\n\t"
      "global_load_async_to_lds_b128 %0, %1, off offset:16\n\t"
      "global_load_async_to_lds_b128 %0, %1, off offset:32\n\t"
      "global_load_async_to_lds_b128 %0, %1, off offset:48"
      :: "v"(lds), "v"(ga)
      : "memory");
}
__device__ __forceinline__ void wait_async0() {
  asm volatile("s_wait_asynccnt 0" ::: "memory");
}

__device__ __forceinline__ v16bf combine16(v8bf lo, v8bf hi) {
  v16bf r;
#pragma unroll
  for (int i = 0; i < 8; ++i) { r[i] = lo[i]; r[i + 8] = hi[i]; }
  return r;
}

__global__ __launch_bounds__(256)
void gemm_bf16_wmma_kernel(const __bf16* __restrict__ A,   // [M,K] bf16
                           const __bf16* __restrict__ Bw,  // [N,K] bf16
                           float*        __restrict__ C) { // [M,N] f32
  __shared__ __align__(16) __bf16 sA[2][BM * LDT];   // 2 x 18KB
  __shared__ __align__(16) __bf16 sB[2][BN * LDT];   // 2 x 18KB

  const int m0   = blockIdx.x * BM;
  const int n0   = blockIdx.y * BN;
  const int t    = threadIdx.x;
  const int lane = t & 31;
  const int wave = t >> 5;             // 0..7
  const int wm   = (wave & 3) * 32;    // 4 waves along M
  const int wn   = (wave >> 2) * 64;   // 2 waves along N
  const int lr   = lane & 15;
  const int lh   = lane >> 4;          // lane half (0/1)

  // Tile fill: 128 rows x 64 cols bf16 (16KB/operand). Thread t: 64B of a row.
  const int lrow  = t >> 1;            // 0..127
  const int lcolE = (t & 1) << 5;      // element col: 0 or 32
  const __bf16* gA = A  + (size_t)(m0 + lrow) * Kdim + lcolE;
  const __bf16* gB = Bw + (size_t)(n0 + lrow) * Kdim + lcolE;

  // 32-bit LDS byte addresses (low 32 bits of the generic pointer).
  const unsigned int bufBytes = (unsigned int)(BM * LDT * sizeof(__bf16));
  const unsigned int la0 = (unsigned int)(size_t)&sA[0][lrow * LDT + lcolE];
  const unsigned int lb0 = (unsigned int)(size_t)&sB[0][lrow * LDT + lcolE];
  const unsigned int ldsA[2] = { la0, la0 + bufBytes };
  const unsigned int ldsB[2] = { lb0, lb0 + bufBytes };

  v8f acc[2][4];
#pragma unroll
  for (int ma = 0; ma < 2; ++ma)
#pragma unroll
    for (int nb = 0; nb < 4; ++nb) acc[ma][nb] = {};

  // Preload K-tile 0 through the async engine.
  async_fill64(ldsA[0], gA);
  async_fill64(ldsB[0], gB);
  wait_async0();
  __syncthreads();

  constexpr int KT = Kdim / BK;        // 64 tile steps
  for (int kt = 0; kt < KT; ++kt) {
    const int buf = kt & 1;
    if (kt + 1 < KT) {                 // fill next tile while computing this one
      const int kk = (kt + 1) * BK;
      async_fill64(ldsA[buf ^ 1], gA + kk);
      async_fill64(ldsB[buf ^ 1], gB + kk);
      if (kt + 2 < KT) {               // warm L2 ahead of the async engine
        __builtin_prefetch(gA + (size_t)(kt + 2) * BK, 0, 1);
        __builtin_prefetch(gB + (size_t)(kt + 2) * BK, 0, 1);
      }
    }

#pragma unroll
    for (int ks = 0; ks < 2; ++ks) {   // two WMMA K-steps per LDS tile
      // A frag: lane holds x[m=lr][kb+0..7 , kb+16..23], kb = ks*32 + lh*8
      v16bf af[2];
#pragma unroll
      for (int ma = 0; ma < 2; ++ma) {
        const int base = (wm + ma * 16 + lr) * LDT + ks * 32 + (lh << 3);
        af[ma] = combine16(*(const v8bf*)(&sA[buf][base]),
                           *(const v8bf*)(&sA[buf][base + 16]));
      }
      // B frag: lane holds w[n=lr][kb+0..7 , kb+16..23], kb = ks*32 + lh*8
      v16bf bfr[4];
#pragma unroll
      for (int nb = 0; nb < 4; ++nb) {
        const int base = (wn + nb * 16 + lr) * LDT + ks * 32 + (lh << 3);
        bfr[nb] = combine16(*(const v8bf*)(&sB[buf][base]),
                            *(const v8bf*)(&sB[buf][base + 16]));
      }
#pragma unroll
      for (int ma = 0; ma < 2; ++ma)
#pragma unroll
        for (int nb = 0; nb < 4; ++nb)
          acc[ma][nb] = __builtin_amdgcn_wmma_f32_16x16x32_bf16(
              false, af[ma], false, bfr[nb], (short)0, acc[ma][nb], false, false);
    }

    if (kt + 1 < KT) wait_async0();    // next tile landed in LDS
    __syncthreads();                   // publish across waves / guard reuse
  }

  // C layout: lane L, VGPR r -> m = r + (L/16)*8, n = L%16
#pragma unroll
  for (int ma = 0; ma < 2; ++ma) {
#pragma unroll
    for (int nb = 0; nb < 4; ++nb) {
      const int mb = m0 + wm + ma * 16 + (lh << 3);
      const int n  = n0 + wn + nb * 16 + lr;
#pragma unroll
      for (int r = 0; r < 8; ++r)
        C[(size_t)(mb + r) * Ndim + n] = acc[ma][nb][r];
    }
  }
}

// ------------------------------- launcher -----------------------------------
extern "C" void kernel_launch(void* const* d_in, const int* in_sizes, int n_in,
                              void* d_out, int out_size, void* d_ws, size_t ws_size,
                              hipStream_t stream) {
  const float* x  = (const float*)d_in[0];
  const float* cb = (const float*)d_in[1];
  const float* sc = (const float*)d_in[2];
  const int*   ix = (const int*)d_in[3];
  float* out = (float*)d_out;

  // workspace: [ x.bf16 : M*K*2 = 64MiB ][ w.bf16 : N*K*2 = 128MiB ]
  __bf16* xb = (__bf16*)d_ws;
  __bf16* wb = (__bf16*)((char*)d_ws + (size_t)Mdim * Kdim * 2);

  cvt_x_kernel<<<(Mdim * Kdim / 8) / 256, 256, 0, stream>>>(x, xb);
  dequant_w_kernel<<<(OUTsz * Gn) / 256, 256, 0, stream>>>(cb, sc, ix, wb);

  dim3 grid(Mdim / BM, Ndim / BN);     // x-fastest: weight panel reuse in L2
  gemm_bf16_wmma_kernel<<<grid, 256, 0, stream>>>(xb, wb, out);
}
